// RingAttractorNetwork_66348654788912
// MI455X (gfx1250) — matrix operand order repaired and checked by reading, compiled, not run on hardware
//
#include <hip/hip_runtime.h>
#include <hip/hip_bf16.h>
#include <stdint.h>

// ---------------------------------------------------------------------------
// Ring attractor network, CDNA5 (gfx1250) WMMA implementation.
//   N_EXC=1024, N_INH=256, BATCH=2048, 10 steps.
//   GEMMs on v_wmma_f32_16x16x32_bf16 with 32x32 register tiles per wave
//   (2 A-frags x 2 B-frags -> 4 WMMAs per K-chunk, 2 loads/WMMA).
// ---------------------------------------------------------------------------

#define N_EXC  1024
#define N_INH  256
#define BATCH  2048
#define NSTEPS 10          // reference STEPS (device scalar not readable in capture)
#define DT_TAU_E 0.01f     // DT/TAU_E = 0.1/10
#define DT_TAU_I 0.02f     // DT/TAU_I = 0.1/5

typedef __attribute__((ext_vector_type(16))) __bf16 v16bf;
typedef __attribute__((ext_vector_type(8)))  __bf16 v8bf;
typedef __attribute__((ext_vector_type(8)))  float  v8f;

static __device__ inline __bf16 f32_to_bf16(float f) {
    union { float f; uint32_t u; } un; un.f = f;
    uint32_t u = un.u;
    uint32_t lsb = (u >> 16) & 1u;           // round-to-nearest-even
    u += 0x7fffu + lsb;
    unsigned short h = (unsigned short)(u >> 16);
    return __builtin_bit_cast(__bf16, h);
}

static __device__ inline v16bf combine8(v8bf lo, v8bf hi) {
    return __builtin_shufflevector(lo, hi, 0,1,2,3,4,5,6,7,8,9,10,11,12,13,14,15);
}

// A-fragment for 16-bit 16x32 A: lane half 0 -> K{0-15}, half 1 -> K{16-31}, contiguous.
static __device__ inline v16bf loadA(const __bf16* __restrict__ row, int kc, int aoff) {
    return combine8(*(const v8bf*)(row + kc + 2 * aoff),
                    *(const v8bf*)(row + kc + 2 * aoff + 8));
}
// B-fragment for 16-bit 32x16 B: lane half selects K group {0-15}/{16-31}, contiguous.
static __device__ inline v16bf loadB(const __bf16* __restrict__ row, int kc, int boff) {
    return combine8(*(const v8bf*)(row + kc + boff),
                    *(const v8bf*)(row + kc + boff + 8));
}

static __device__ inline v8f wmma_bf16(v16bf A, v16bf B, v8f C) {
    return __builtin_amdgcn_wmma_f32_16x16x32_bf16(false, A, false, B,
                                                   (short)0, C, false, false);
}

// PCG-style hash for deterministic, capture-safe noise.
static __device__ inline unsigned pcg_hash(unsigned x) {
    x = x * 747796405u + 2891336453u;
    unsigned w = ((x >> ((x >> 28) + 4u)) ^ x) * 277803737u;
    return (w >> 22) ^ w;
}

// Centered Poisson sample (Knuth, small rate => ~1 iteration).
static __device__ inline float poisson_centered(unsigned seed, float rate, float expNegRate) {
    float p = 1.0f;
    int k = -1;
    unsigned s = seed;
    do {
        ++k;
        s = pcg_hash(s);
        float u = (float)(s >> 8) * (1.0f / 16777216.0f);
        p *= u;
    } while (p > expNegRate && k < 16);
    return (float)k - rate;
}

// ---------------------------------------------------------------------------
// W_EE ring weights: one block per row i. Gaussian in wrapped angular distance,
// row-normalized (sum includes diagonal), diagonal zeroed after normalization.
// Stored bf16 row-major [i][j] == WMMA-B column-contiguous layout.
// ---------------------------------------------------------------------------
__global__ __launch_bounds__(256) void ring_wee_kernel(const float* __restrict__ sigma_p,
                                                       __bf16* __restrict__ Wee) {
    __shared__ float wrow[N_EXC];
    __shared__ float partial[256];
    const int i = blockIdx.x;
    const float sigma    = *sigma_p;
    const float astep    = 6.28318530718f / (float)(N_EXC - 1);  // linspace incl. endpoints
    const float ai       = (float)i * astep;
    const float half_is2 = 0.5f / (sigma * sigma);

    float sum = 0.0f;
    for (int j = threadIdx.x; j < N_EXC; j += 256) {
        float d = (float)j * astep - ai;
        d = d - 6.28318530718f * rintf(d * 0.15915494309f);      // wrap to (-pi, pi]
        float w = __expf(-d * d * half_is2);
        wrow[j] = w;
        sum += w;
    }
    partial[threadIdx.x] = sum;
    __syncthreads();
    for (int s = 128; s > 0; s >>= 1) {
        if (threadIdx.x < (unsigned)s) partial[threadIdx.x] += partial[threadIdx.x + s];
        __syncthreads();
    }
    const float inv = 1.0f / partial[0];
    for (int j = threadIdx.x; j < N_EXC; j += 256) {
        float w = (j == i) ? 0.0f : wrow[j] * inv;
        Wee[(size_t)i * N_EXC + j] = f32_to_bf16(w);
    }
}

// ---------------------------------------------------------------------------
// Pre-transpose W_IE (256x1024 -> [i][k] 1024x256 bf16) and
//               W_EI (1024x256 -> [ki][e] 256x1024 bf16).
// ---------------------------------------------------------------------------
__global__ __launch_bounds__(256) void prep_weights_kernel(const float* __restrict__ W_EI,
                                                           const float* __restrict__ W_IE,
                                                           __bf16* __restrict__ WeiT,
                                                           __bf16* __restrict__ WieT) {
    const int idx = blockIdx.x * 256 + threadIdx.x;   // 0 .. 1024*256-1
    if (idx < N_EXC * N_INH) {
        int i = idx >> 8, k = idx & (N_INH - 1);          // WieT[i][k] = W_IE[k][i]
        WieT[idx] = f32_to_bf16(W_IE[(size_t)k * N_EXC + i]);
        int ki = idx >> 10, e = idx & (N_EXC - 1);        // WeiT[ki][e] = W_EI[e][ki]
        WeiT[idx] = f32_to_bf16(W_EI[(size_t)e * N_INH + ki]);
    }
}

// ---------------------------------------------------------------------------
// State init: r_e = h (f32 + bf16 mirror), r_i = 0.
// ---------------------------------------------------------------------------
__global__ __launch_bounds__(256) void init_state_kernel(const float* __restrict__ h,
                                                         float* __restrict__ reF,
                                                         __bf16* __restrict__ reH,
                                                         float* __restrict__ riF,
                                                         __bf16* __restrict__ riH) {
    const int idx = blockIdx.x * 256 + threadIdx.x;
    if (idx < BATCH * N_EXC) {
        float v = h[idx];
        reF[idx] = v;
        reH[idx] = f32_to_bf16(v);
    }
    if (idx < BATCH * N_INH) {
        riF[idx] = 0.0f;
        riH[idx] = f32_to_bf16(0.0f);
    }
}

// ---------------------------------------------------------------------------
// stepE: input_e = g_ee * (r_e @ Wee^T) - g_ie * (r_i @ W_IE) + g_in * ext (+noise)
//        r_e' = relu(r_e + DT/TAU_E * (-r_e + relu(input_e)))
// One wave32 per 32x32 output tile: 2 A-frags x 2 B-frags -> 4 WMMAs / K-chunk.
// Tiles = 64 x 32 = 2048 waves = 256 blocks x 8 waves.
// ---------------------------------------------------------------------------
__global__ __launch_bounds__(256) void stepE_kernel(const float* __restrict__ reF_in,
                                                    const __bf16* __restrict__ reH_in,
                                                    const __bf16* __restrict__ riH_in,
                                                    const __bf16* __restrict__ Wee,   // [1024][1024]
                                                    const __bf16* __restrict__ WieT,  // [1024][256]
                                                    const float* __restrict__ ext,
                                                    const float* __restrict__ g_ee_p,
                                                    const float* __restrict__ g_ie_p,
                                                    const float* __restrict__ g_in_p,
                                                    const float* __restrict__ nre_p,
                                                    float* __restrict__ reF_out,
                                                    __bf16* __restrict__ reH_out,
                                                    int step) {
    const int wave  = (blockIdx.x * 256 + threadIdx.x) >> 5;
    const int lane  = threadIdx.x & 31;
    const int bt    = wave / (N_EXC / 32);    // 0..63   (batch tile of 32)
    const int it    = wave % (N_EXC / 32);    // 0..31   (exc tile of 32)
    const int half  = lane >> 4;              // 0|1
    const int lrow  = lane & 15;
    const int aoff  = half * 8;
    const int boff  = half * 16;

    const __bf16* a0 = reH_in + (size_t)(bt * 32 + lrow)      * N_EXC;
    const __bf16* a1 = reH_in + (size_t)(bt * 32 + 16 + lrow) * N_EXC;
    const __bf16* b0 = Wee    + (size_t)(it * 32 + lrow)      * N_EXC;
    const __bf16* b1 = Wee    + (size_t)(it * 32 + 16 + lrow) * N_EXC;

    // --- GEMM1: r_e (bf16) x Wee^T, K = 1024 ---
    v8f acc00 = {}, acc01 = {}, acc10 = {}, acc11 = {};
    for (int kc = 0; kc < N_EXC; kc += 32) {
        v16bf A0 = loadA(a0, kc, aoff);
        v16bf A1 = loadA(a1, kc, aoff);
        v16bf B0 = loadB(b0, kc, boff);
        v16bf B1 = loadB(b1, kc, boff);
        acc00 = wmma_bf16(A0, B0, acc00);
        acc01 = wmma_bf16(A0, B1, acc01);
        acc10 = wmma_bf16(A1, B0, acc10);
        acc11 = wmma_bf16(A1, B1, acc11);
    }

    // --- GEMM2: r_i (bf16) x W_IE, K = 256 ---
    const __bf16* c0 = riH_in + (size_t)(bt * 32 + lrow)      * N_INH;
    const __bf16* c1 = riH_in + (size_t)(bt * 32 + 16 + lrow) * N_INH;
    const __bf16* d0 = WieT   + (size_t)(it * 32 + lrow)      * N_INH;
    const __bf16* d1 = WieT   + (size_t)(it * 32 + 16 + lrow) * N_INH;
    v8f ie00 = {}, ie01 = {}, ie10 = {}, ie11 = {};
    for (int kc = 0; kc < N_INH; kc += 32) {
        v16bf A0 = loadA(c0, kc, aoff);
        v16bf A1 = loadA(c1, kc, aoff);
        v16bf B0 = loadB(d0, kc, boff);
        v16bf B1 = loadB(d1, kc, boff);
        ie00 = wmma_bf16(A0, B0, ie00);
        ie01 = wmma_bf16(A0, B1, ie01);
        ie10 = wmma_bf16(A1, B0, ie10);
        ie11 = wmma_bf16(A1, B1, ie11);
    }

    const float g_ee = *g_ee_p, g_ie = *g_ie_p, g_in = *g_in_p, nre = *nre_p;
    const float expn = __expf(-nre);

    // C/D layout: element r -> (M = r + 8*half, N = lrow) within each 16x16 subtile.
    const v8f* ee[4] = { &acc00, &acc01, &acc10, &acc11 };
    const v8f* ie[4] = { &ie00,  &ie01,  &ie10,  &ie11  };
#pragma unroll
    for (int t = 0; t < 4; ++t) {
        const int mi = t >> 1, ni = t & 1;
        const int row0 = bt * 32 + mi * 16 + half * 8;
        const int col  = it * 32 + ni * 16 + lrow;
#pragma unroll
        for (int r = 0; r < 8; ++r) {
            const int b   = row0 + r;
            const int idx = b * N_EXC + col;
            float inp = g_ee * (*ee[t])[r] - g_ie * (*ie[t])[r] + g_in * ext[idx];
            unsigned seed = (unsigned)idx * 2654435761u ^ ((unsigned)step * 0x9E3779B9u) ^ 0xE1234567u;
            inp += poisson_centered(seed, nre, expn);
            float inr = inp > 0.0f ? inp : 0.0f;
            float old = reF_in[idx];
            float v = old + DT_TAU_E * (inr - old);
            v = v > 0.0f ? v : 0.0f;
            reF_out[idx] = v;
            reH_out[idx] = f32_to_bf16(v);
        }
    }
}

// ---------------------------------------------------------------------------
// stepI: input_i = g_ei * (r_e @ W_EI) (+noise)
//        r_i' = relu(r_i + DT/TAU_I * (-r_i + relu(input_i)))
// 32x32 tiles: 64 x 8 = 512 waves = 64 blocks x 8 waves.
// ---------------------------------------------------------------------------
__global__ __launch_bounds__(256) void stepI_kernel(const __bf16* __restrict__ reH_in,
                                                    const float* __restrict__ riF_in,
                                                    const __bf16* __restrict__ WeiT,  // [256][1024]
                                                    const float* __restrict__ g_ei_p,
                                                    const float* __restrict__ nri_p,
                                                    float* __restrict__ riF_out,
                                                    __bf16* __restrict__ riH_out,
                                                    int step) {
    const int wave  = (blockIdx.x * 256 + threadIdx.x) >> 5;
    const int lane  = threadIdx.x & 31;
    const int bt    = wave / (N_INH / 32);    // 0..63
    const int kt    = wave % (N_INH / 32);    // 0..7
    const int half  = lane >> 4;
    const int lrow  = lane & 15;
    const int aoff  = half * 8;
    const int boff  = half * 16;

    const __bf16* a0 = reH_in + (size_t)(bt * 32 + lrow)      * N_EXC;
    const __bf16* a1 = reH_in + (size_t)(bt * 32 + 16 + lrow) * N_EXC;
    const __bf16* b0 = WeiT   + (size_t)(kt * 32 + lrow)      * N_EXC;
    const __bf16* b1 = WeiT   + (size_t)(kt * 32 + 16 + lrow) * N_EXC;

    v8f acc00 = {}, acc01 = {}, acc10 = {}, acc11 = {};
    for (int kc = 0; kc < N_EXC; kc += 32) {
        v16bf A0 = loadA(a0, kc, aoff);
        v16bf A1 = loadA(a1, kc, aoff);
        v16bf B0 = loadB(b0, kc, boff);
        v16bf B1 = loadB(b1, kc, boff);
        acc00 = wmma_bf16(A0, B0, acc00);
        acc01 = wmma_bf16(A0, B1, acc01);
        acc10 = wmma_bf16(A1, B0, acc10);
        acc11 = wmma_bf16(A1, B1, acc11);
    }

    const float g_ei = *g_ei_p, nri = *nri_p;
    const float expn = __expf(-nri);
    const v8f* ac[4] = { &acc00, &acc01, &acc10, &acc11 };
#pragma unroll
    for (int t = 0; t < 4; ++t) {
        const int mi = t >> 1, ni = t & 1;
        const int row0 = bt * 32 + mi * 16 + half * 8;
        const int col  = kt * 32 + ni * 16 + lrow;
#pragma unroll
        for (int r = 0; r < 8; ++r) {
            const int b   = row0 + r;
            const int idx = b * N_INH + col;
            float inp = g_ei * (*ac[t])[r];
            unsigned seed = (unsigned)idx * 2246822519u ^ ((unsigned)step * 0x9E3779B9u) ^ 0x1B873593u;
            inp += poisson_centered(seed, nri, expn);
            float inr = inp > 0.0f ? inp : 0.0f;
            float old = riF_in[idx];
            float v = old + DT_TAU_I * (inr - old);
            v = v > 0.0f ? v : 0.0f;
            riF_out[idx] = v;
            riH_out[idx] = f32_to_bf16(v);
        }
    }
}

// ---------------------------------------------------------------------------
// Host launcher.
// ---------------------------------------------------------------------------
extern "C" void kernel_launch(void* const* d_in, const int* in_sizes, int n_in,
                              void* d_out, int out_size, void* d_ws, size_t ws_size,
                              hipStream_t stream) {
    const float* ext     = (const float*)d_in[0];   // [2048][1024]
    const float* h       = (const float*)d_in[1];   // [2048][1024]
    const float* sigma   = (const float*)d_in[2];
    const float* W_EI    = (const float*)d_in[3];   // [1024][256]
    const float* W_IE    = (const float*)d_in[4];   // [256][1024]
    const float* g_ee    = (const float*)d_in[5];
    const float* g_ei    = (const float*)d_in[6];
    const float* g_ie    = (const float*)d_in[7];
    const float* g_in    = (const float*)d_in[8];
    const float* nre     = (const float*)d_in[9];
    const float* nri     = (const float*)d_in[10];
    // d_in[11] = steps (device int); fixed to NSTEPS=10 per reference module.

    // Workspace carving (256B aligned slices).
    char* w = (char*)d_ws;
    size_t off = 0;
    auto carve = [&](size_t bytes) -> char* {
        char* p = w + off;
        off = (off + bytes + 255) & ~(size_t)255;
        return p;
    };
    __bf16* Wee   = (__bf16*)carve((size_t)N_EXC * N_EXC * 2);
    __bf16* WieT  = (__bf16*)carve((size_t)N_EXC * N_INH * 2);
    __bf16* WeiT  = (__bf16*)carve((size_t)N_INH * N_EXC * 2);
    float*  reF[2]; __bf16* reH[2]; float* riF[2]; __bf16* riH[2];
    for (int i = 0; i < 2; ++i) reF[i] = (float*) carve((size_t)BATCH * N_EXC * 4);
    for (int i = 0; i < 2; ++i) reH[i] = (__bf16*)carve((size_t)BATCH * N_EXC * 2);
    for (int i = 0; i < 2; ++i) riF[i] = (float*) carve((size_t)BATCH * N_INH * 4);
    for (int i = 0; i < 2; ++i) riH[i] = (__bf16*)carve((size_t)BATCH * N_INH * 2);
    (void)ws_size; (void)in_sizes; (void)n_in; (void)out_size;

    // One-time prep (re-run every call: deterministic, capture-safe).
    ring_wee_kernel<<<N_EXC, 256, 0, stream>>>(sigma, Wee);
    prep_weights_kernel<<<(N_EXC * N_INH) / 256, 256, 0, stream>>>(W_EI, W_IE, WeiT, WieT);
    init_state_kernel<<<(BATCH * N_EXC) / 256, 256, 0, stream>>>(h, reF[0], reH[0], riF[0], riH[0]);

    const int gridE = (BATCH / 32) * (N_EXC / 32) * 32 / 256;  // 256 blocks
    const int gridI = (BATCH / 32) * (N_INH / 32) * 32 / 256;  // 64 blocks

    int cur = 0;
    for (int s = 0; s < NSTEPS; ++s) {
        const int nxt = cur ^ 1;
        float* reOutF = (s == NSTEPS - 1) ? (float*)d_out : reF[nxt];
        stepE_kernel<<<gridE, 256, 0, stream>>>(reF[cur], reH[cur], riH[cur],
                                                Wee, WieT, ext,
                                                g_ee, g_ie, g_in, nre,
                                                reOutF, reH[nxt], s);
        if (s < NSTEPS - 1) {
            stepI_kernel<<<gridI, 256, 0, stream>>>(reH[cur], riF[cur], WeiT,
                                                    g_ei, nri,
                                                    riF[nxt], riH[nxt], s);
        }
        cur = nxt;
    }
}
